// Deformable_Pooling_1D_2551210574030
// MI455X (gfx1250) — matrix operand order repaired and checked by reading, not compiled, not run on hardware
//
#include <hip/hip_runtime.h>
#include <hip/hip_bf16.h>
#include <stddef.h>
#include <stdint.h>

// ---------------- problem constants (fixed by setup_inputs) ----------------
static constexpr int B  = 8;
static constexpr int C  = 256;
static constexpr int H  = 128;
static constexpr int Wd = 128;
static constexpr int HW = H * Wd;
static constexpr int HP = H + 2;                       // padded (halo) dims
static constexpr int WP = Wd + 2;
static constexpr int TILES_PER_ROW = Wd / 16;          // 8
static constexpr int NTILES = B * H * TILES_PER_ROW;   // 8192 (16-pixel tiles)
static constexpr int NTAPS  = 9;                       // 3x3
static constexpr int KCHUNKS = C / 32;                 // 8  (K=32 per WMMA)
static constexpr int NFRAG  = NTAPS * KCHUNKS;         // 72 B-fragments
static constexpr int WF_BYTES = NFRAG * 32 * 16 * 2;   // 73,728 B packed weights

// ---------------- workspace layout ----------------
static constexpr size_t XT_OFF  = 0;                                  // NHWC bf16 x, with halo
static constexpr size_t XT_SZ   = (size_t)B * HP * WP * C * 2;        // 69,222,400
static constexpr size_t WF_OFF  = XT_OFF + XT_SZ;                     // packed weight fragments
static constexpr size_t IDX_OFF = WF_OFF + WF_BYTES;                  // per-pixel gather base idx
static constexpr size_t IDX_SZ  = (size_t)B * HW * 4;                 // 524,288
static constexpr size_t WGT_OFF = IDX_OFF + IDX_SZ;                   // per-pixel 4 bilinear wts
// total ~71.9 MB

// ---------------- vector types ----------------
typedef __bf16        v16bf __attribute__((ext_vector_type(16)));
typedef float         v8f   __attribute__((ext_vector_type(8)));
typedef unsigned int  u32x4 __attribute__((ext_vector_type(4)));
typedef int           i32x4 __attribute__((ext_vector_type(4)));
typedef int           i32x8 __attribute__((ext_vector_type(8)));

union AFrag { v16bf v; u32x4 q[2]; };   // 32B A-operand fragment (16 bf16 / lane)

static __device__ __forceinline__ __bf16 f2bf(float f) {
    unsigned u = __builtin_bit_cast(unsigned, f);
    u += 0x7FFFu + ((u >> 16) & 1u);            // round-to-nearest-even
    unsigned short s = (unsigned short)(u >> 16);
    return __builtin_bit_cast(__bf16, s);
}

// ============================================================================
// Kernel 1: pack conv weights into WMMA B-fragments (bf16, N padded 2 -> 16).
// B-matrix 32x16 bf16, this code's contiguous k order (A matches): lane L holds column n = L&15,
// K-half selected by L>>4; element j of the lane's v16bf is k = j + 16*(L>>4).
// ============================================================================
__global__ void wprep_kernel(const float* __restrict__ w_conv,
                             __bf16* __restrict__ wf) {
    const int tap = blockIdx.x / KCHUNKS;
    const int cc  = blockIdx.x % KCHUNKS;
    const int L   = threadIdx.x;                 // 0..31
    const int kh  = tap / 3, kw = tap % 3;
    const int n   = L & 15;
    const int kbase = (L >> 4) * 16;
    __bf16* dst = wf + ((size_t)(tap * KCHUNKS + cc) * 32 + L) * 16;
#pragma unroll
    for (int j = 0; j < 16; ++j) {
        const int ch = cc * 32 + kbase + j;      // input channel
        float v = 0.0f;
        if (n < 2) v = w_conv[((n * C + ch) * 3 + kh) * 3 + kw];
        dst[j] = f2bf(v);
    }
}

// ============================================================================
// Kernel 2a: zero the halo ring of the padded NHWC buffer (16B per thread).
// ============================================================================
static constexpr int HALO_SLOTS = 2 * WP + 2 * H;      // 516 border pixels / image
__global__ void halo_zero_kernel(__bf16* __restrict__ xTp) {
    const unsigned t = blockIdx.x * blockDim.x + threadIdx.x;
    if (t >= (unsigned)B * HALO_SLOTS * 32) return;
    const unsigned chunk = t & 31;               // 32 x 16B = 512B = 256 ch
    const unsigned s = (t >> 5) % HALO_SLOTS;
    const unsigned b = (t >> 5) / HALO_SLOTS;
    int row, col;
    if (s < (unsigned)WP)          { row = 0;            col = (int)s; }
    else if (s < (unsigned)(2*WP)) { row = HP - 1;       col = (int)s - WP; }
    else if (s < (unsigned)(2*WP + H)) { row = (int)s - 2*WP + 1;  col = 0; }
    else                           { row = (int)s - (2*WP + H) + 1; col = WP - 1; }
    u32x4* p = (u32x4*)(xTp + ((size_t)(b * HP + row) * WP + col) * C) + chunk;
    *p = (u32x4)0u;
}

// ============================================================================
// Kernel 2b: NCHW f32 -> padded NHWC bf16 transpose (LDS tiled).
// ============================================================================
__global__ void nhwc_kernel(const float* __restrict__ x,
                            __bf16* __restrict__ xTp) {
    __shared__ float tile[32][33];
    const int hw0 = blockIdx.x * 32;
    const int c0  = blockIdx.y * 32;
    const int b   = blockIdx.z;
    const float* xp = x + (size_t)b * C * HW;
#pragma unroll
    for (int i = 0; i < 4; ++i) {
        const int c = c0 + threadIdx.y + i * 8;
        tile[threadIdx.y + i * 8][threadIdx.x] =
            xp[(size_t)c * HW + hw0 + threadIdx.x];
    }
    __syncthreads();
#pragma unroll
    for (int i = 0; i < 4; ++i) {
        const int hw = hw0 + threadIdx.y + i * 8;
        const int h = hw >> 7, w = hw & (Wd - 1);
        xTp[((size_t)(b * HP + h + 1) * WP + (w + 1)) * C + c0 + threadIdx.x] =
            f2bf(tile[threadIdx.x][threadIdx.y + i * 8]);
    }
}

// ============================================================================
// TDM: broadcast the 72KB packed-weight block into this block's LDS.
// 1-D copy described as D#: data_size=8B, tensor_dim0 = tile_dim0 = 9216.
// ============================================================================
static __device__ __forceinline__ void tdm_load_weights(const __bf16* wf,
                                                        __bf16* swf) {
    if (threadIdx.x == 0) {
        const unsigned lds = (unsigned)(uintptr_t)swf;       // low 32b = LDS addr
        const unsigned long long ga = (unsigned long long)(uintptr_t)wf;
        const unsigned n8 = WF_BYTES / 8;                    // 9216 qwords

        u32x4 g0;
        g0[0] = 1u;                                          // count=1, load
        g0[1] = lds;                                         // lds_addr
        g0[2] = (unsigned)ga;                                // global_addr[31:0]
        g0[3] = ((unsigned)(ga >> 32) & 0x01FFFFFFu) | 0x80000000u; // addr[56:32] | type=2

        i32x8 g1;
        g1[0] = (int)(3u << 16);                             // data_size=8B, mask=0
        g1[1] = (int)((n8 & 0xFFFFu) << 16);                 // tensor_dim0[15:0]
        g1[2] = (int)((n8 >> 16) | (1u << 16));              // tensor_dim0[31:16], tensor_dim1=1
        g1[3] = (int)((n8 & 0xFFFFu) << 16);                 // tile_dim0 = n8
        g1[4] = 1;                                           // tile_dim1 = 1
        g1[5] = (int)n8;                                     // tensor_dim0_stride
        g1[6] = 0;
        g1[7] = 0;

        i32x4 gz = {0, 0, 0, 0};
#if __clang_major__ >= 23
        i32x8 gz8 = {0, 0, 0, 0, 0, 0, 0, 0};
        __builtin_amdgcn_tensor_load_to_lds(g0, g1, gz, gz, gz8, 0);
#else
        __builtin_amdgcn_tensor_load_to_lds(g0, g1, gz, gz, 0);
#endif
    }
    __builtin_amdgcn_s_wait_tensorcnt(0);    // no-op for waves that didn't issue
    __syncthreads();
}

// ============================================================================
// Kernel 3: offset conv as bf16 WMMA GEMM + px/py/bilinear metadata.
// One wave per 16-pixel tile; halo-padded input -> branch-free A loads;
// weights served from LDS (filled by TDM); 72 WMMAs into one v8f.
// ============================================================================
__global__ void conv_offset_kernel(const __bf16* __restrict__ xTp,
                                   const __bf16* __restrict__ wf,
                                   const float* __restrict__ b_conv,
                                   int* __restrict__ idx_out,
                                   float4* __restrict__ wgt_out) {
    __shared__ __bf16 swf[NFRAG * 32 * 16];      // 73,728 B weight fragments
    __shared__ float  sm[8][16][2];              // [wave][m][n<2]

    tdm_load_weights(wf, swf);

    const int tid  = threadIdx.x;
    const int wv   = tid >> 5;                   // wave in block (0..7)
    const int lane = tid & 31;
    const int tile = blockIdx.x * 8 + wv;

    const int w0 = (tile & (TILES_PER_ROW - 1)) * 16;
    const int h  = (tile >> 3) & (H - 1);
    const int b  = tile >> 10;

    const int m    = lane & 15;                  // A-matrix row (pixel in tile)
    const int half = lane >> 4;                  // K-half select

    // lane's base address: padded pixel (b, h, w0+m), channel 16*half.
    // tap (dh,dw) is reached by a compile-time offset (dh*WP + dw)*C.
    const __bf16* abase = xTp + ((size_t)(b * HP + h) * WP + (w0 + m)) * C + 16 * half;
    const v16bf* swfv = (const v16bf*)swf;

    v8f acc = {0.f, 0.f, 0.f, 0.f, 0.f, 0.f, 0.f, 0.f};

#pragma unroll
    for (int tap = 0; tap < NTAPS; ++tap) {
        const int dh = tap / 3, dw = tap % 3;
        const __bf16* src = abase + (size_t)(dh * WP + dw) * C;
#pragma unroll
        for (int cc = 0; cc < KCHUNKS; ++cc) {
            AFrag a;
            a.q[0] = *(const u32x4*)(src + cc * 32);
            a.q[1] = *(const u32x4*)(src + cc * 32 + 8);
            const v16bf bw = swfv[(tap * KCHUNKS + cc) * 32 + lane];
            acc = __builtin_amdgcn_wmma_f32_16x16x32_bf16(
                false, a.v, false, bw, (short)0, acc, false, false);
        }
    }

    // D layout: lane L, reg r -> M = r + 8*(L>>4), N = L&15. Keep N=0,1.
    if ((lane & 15) < 2) {
#pragma unroll
        for (int r = 0; r < 8; ++r)
            sm[wv][r + 8 * half][lane & 15] = acc[r];
    }
    __syncthreads();

    // 128 pixels per block: thread t -> wave t>>4, pixel m = t&15
    if (tid < 128) {
        const int pwv = tid >> 4;
        const int pm  = tid & 15;
        const int ptile = blockIdx.x * 8 + pwv;
        const int pw0 = (ptile & (TILES_PER_ROW - 1)) * 16;
        const int ph  = (ptile >> 3) & (H - 1);
        const int pb  = ptile >> 10;

        const float ox = sm[pwv][pm][0] + b_conv[0];
        const float oy = sm[pwv][pm][1] + b_conv[1];

        float px = (float)ph + ox;
        float py = (float)(pw0 + pm) + oy;
        px = fminf(fmaxf(px, 0.0f), (float)(H - 2));
        py = fminf(fmaxf(py, 0.0f), (float)(Wd - 2));

        const float fx = floorf(px), fy = floorf(py);
        const int qx0 = (int)fx, qy0 = (int)fy;
        const float ax = px - fx, ay = py - fy;

        const float g_lt = (1.0f - ax) * (1.0f - ay);
        const float g_rb = ax * ay;
        const float g_lb = (1.0f - ax) * ay;
        const float g_rt = ax * (1.0f - ay);

        const int pix = (pb * H + ph) * Wd + (pw0 + pm);
        idx_out[pix] = qx0 * Wd + qy0;
        wgt_out[pix] = make_float4(g_lt, g_rb, g_lb, g_rt);
    }
}

// ============================================================================
// Kernel 4: bilinear gather, 8 channels per thread (reuse idx/weights),
// coalesced stores per channel plane, prefetch next plane.
// ============================================================================
static constexpr int CPT = 8;

__global__ void gather_kernel(const float* __restrict__ x,
                              const int* __restrict__ idx_in,
                              const float4* __restrict__ wgt_in,
                              float* __restrict__ out) {
    const unsigned t = blockIdx.x * blockDim.x + threadIdx.x;
    const int w  =  t        & (Wd - 1);
    const int h  = (t >> 7)  & (H - 1);
    const int cg = (t >> 14) & (C / CPT - 1);
    const int b  =  t >> 19;

    const int pix = (b * H + h) * Wd + w;
    const int    i0 = idx_in[pix];
    const float4 g  = wgt_in[pix];

    const int o00 = i0;
    const int o01 = i0 + 1;
    const int o10 = i0 + Wd;
    const int o11 = i0 + Wd + 1;

    const float* base  = x   + (size_t)(b * C + cg * CPT) * HW;
    float*       obase = out + (size_t)(b * C + cg * CPT) * HW + h * Wd + w;

#pragma unroll
    for (int c = 0; c < CPT; ++c) {
        if (c + 1 < CPT)
            __builtin_prefetch(base + HW + o00, 0, 1);   // global_prefetch_b8
        const float v = g.x * base[o00] + g.y * base[o11]
                      + g.z * base[o01] + g.w * base[o10];
        obase[(size_t)c * HW] = v;
        base += HW;
    }
}

// ============================================================================
// launch
// ============================================================================
extern "C" void kernel_launch(void* const* d_in, const int* in_sizes, int n_in,
                              void* d_out, int out_size, void* d_ws, size_t ws_size,
                              hipStream_t stream) {
    (void)in_sizes; (void)n_in; (void)out_size; (void)ws_size;

    const float* x      = (const float*)d_in[0];   // [8,256,128,128]
    const float* w_conv = (const float*)d_in[1];   // [2,256,3,3]
    const float* b_conv = (const float*)d_in[2];   // [2]
    float*       out    = (float*)d_out;           // [8,256,128,128,1]

    char* ws = (char*)d_ws;
    __bf16* xTp  = (__bf16*)(ws + XT_OFF);
    __bf16* wf   = (__bf16*)(ws + WF_OFF);
    int*    idxb = (int*)   (ws + IDX_OFF);
    float4* wgtb = (float4*)(ws + WGT_OFF);

    // 1) pack weights into WMMA B fragments
    wprep_kernel<<<NFRAG, 32, 0, stream>>>(w_conv, wf);

    // 2) halo zero + NCHW f32 -> padded NHWC bf16
    const unsigned halo_threads = (unsigned)B * HALO_SLOTS * 32;
    halo_zero_kernel<<<(halo_threads + 255) / 256, 256, 0, stream>>>(xTp);
    nhwc_kernel<<<dim3(HW / 32, C / 32, B), dim3(32, 8), 0, stream>>>(x, xTp);

    // 3) offset conv (WMMA + TDM weight broadcast) + bilinear metadata
    conv_offset_kernel<<<NTILES / 8, 256, 0, stream>>>(
        xTp, wf, b_conv, idxb, wgtb);

    // 4) gather / interpolate
    const unsigned nthreads = (unsigned)B * HW * (C / CPT);   // 4,194,304
    gather_kernel<<<nthreads / 256, 256, 0, stream>>>(x, idxb, wgtb, out);
}
